// LSTM_26396869001416
// MI455X (gfx1250) — compile-verified, matrix operand order checked by reading, not run on hardware
//
#include <hip/hip_runtime.h>

// ---------------------------------------------------------------------------
// LSTM for MI455X (gfx1250, wave32, WMMA).
//   Phase 1: xw[s][b][4H] = x @ [Wf|Wi|Wo|Wc] + b   (bf16 WMMA, f32 accum)
//   Phase 2: persistent scan kernel, 32 WGs, one 16-col H slice each.
//            U (2MB) + h (64KB) live in the 192MB L2; one grid barrier/step.
// ---------------------------------------------------------------------------

typedef __bf16 bf16;
typedef __attribute__((ext_vector_type(16))) __bf16 v16bf;
typedef __attribute__((ext_vector_type(8)))  float  v8f;
typedef __attribute__((ext_vector_type(4)))  __bf16 bf16x4;

#define B_   64
#define S_   512
#define I_   512
#define H_   512
#define G4   2048      // 4*H
#define KT   16        // 512 / 32 k-tiles
#define NWG  32        // scan workgroups: 512 H-cols / 16

union FragBF { v16bf v; float4 q[2]; };

// A fragment: 16x32 bf16, row-major source with leading dim `ld` (elements).
// ISA layout: lanes 0-15 row m hold K kb..kb+7 (q0) and kb+16..kb+23 (q1),
// lanes 16-31 kb=8.  -> two global/ds b128 loads per lane.
__device__ inline v16bf load_a_frag(const bf16* tile, int ld, int lane) {
  const bf16* p = tile + (lane & 15) * ld + ((lane >> 4) << 3);
  FragBF f;
  f.q[0] = *(const float4*)(p);
  f.q[1] = *(const float4*)(p + 16);
  return f.v;
}

// B fragment: pre-swizzled 32x16 tile, 512 bf16 elems, lane-contiguous 32B.
__device__ inline v16bf load_b_frag(const bf16* tile, int lane) {
  const bf16* p = tile + lane * 16;
  FragBF f;
  f.q[0] = *(const float4*)(p);
  f.q[1] = *(const float4*)(p + 8);
  return f.v;
}

__device__ inline v8f wmma_bf16(v16bf a, v16bf b, v8f c) {
  return __builtin_amdgcn_wmma_f32_16x16x32_bf16(false, a, false, b,
                                                 (short)0, c, false, false);
}

__device__ inline float sigmoidf_(float x) { return 1.0f / (1.0f + __expf(-x)); }

__device__ inline void grid_barrier(unsigned* bar, unsigned target) {
  __threadfence();
  __syncthreads();
  if (threadIdx.x == 0) {
    atomicAdd(bar, 1u);
    volatile unsigned* vb = (volatile unsigned*)bar;
    while (*vb < target) __builtin_amdgcn_s_sleep(1);
  }
  __syncthreads();
  __threadfence();
}

// --------------------------- init (per call) -------------------------------
__global__ __launch_bounds__(256) void k_init(bf16* h0, unsigned* bar) {
  float4 z = {0.f, 0.f, 0.f, 0.f};
  for (int i = threadIdx.x; i < (B_ * H_ * 2) / 16; i += 256)
    ((float4*)h0)[i] = z;
  if (threadIdx.x == 0) *bar = 0u;
}

// --------------------------- x -> bf16 -------------------------------------
__global__ __launch_bounds__(256) void k_cvt_x(const float* __restrict__ x,
                                               bf16* __restrict__ xbf) {
  size_t i = ((size_t)blockIdx.x * 256 + threadIdx.x) * 4;
  float4 v = *(const float4*)(x + i);
  bf16x4 o;
  o[0] = (bf16)v.x; o[1] = (bf16)v.y; o[2] = (bf16)v.z; o[3] = (bf16)v.w;
  *(bf16x4*)(xbf + i) = o;
}

// ------------------- W/U -> bf16, swizzled into B-fragment tiles -----------
// dst tile (ntile, ktile): 512 elems; element (kk,n) at lane*16+j where
//   lane = n + ((kk>>3)&1)*16 ,  j = (kk&7) + ((kk>>4)&1)*8
__global__ __launch_bounds__(256) void k_cvt_w(
    const float* __restrict__ Wf, const float* __restrict__ Wi,
    const float* __restrict__ Wo, const float* __restrict__ Wc,
    const float* __restrict__ Uf, const float* __restrict__ Ui,
    const float* __restrict__ Uo, const float* __restrict__ Uc,
    bf16* __restrict__ Wsw, bf16* __restrict__ Usw) {
  const float* src;
  switch (blockIdx.y) {
    case 0: src = Wf; break; case 1: src = Wi; break;
    case 2: src = Wo; break; case 3: src = Wc; break;
    case 4: src = Uf; break; case 5: src = Ui; break;
    case 6: src = Uo; break; default: src = Uc; break;
  }
  bf16* dst = (blockIdx.y < 4) ? Wsw : Usw;
  int gate = blockIdx.y & 3;
  int e = blockIdx.x * 256 + threadIdx.x;   // 0..262143 = k*512+col
  int k = e >> 9, col = e & 511;
  float v = src[e];
  int nglob = (gate << 9) + col;
  int ntile = nglob >> 4, n = nglob & 15;
  int ktile = k >> 5,  kk = k & 31;
  int lane = n + (((kk >> 3) & 1) << 4);
  int j = (kk & 7) + (((kk >> 4) & 1) << 3);
  dst[((size_t)(ntile * KT + ktile) << 9) + (lane << 4) + j] = (bf16)v;
}

__global__ __launch_bounds__(256) void k_cvt_b(const float* __restrict__ b0,
                                               const float* __restrict__ b1,
                                               const float* __restrict__ b2,
                                               const float* __restrict__ b3,
                                               float* __restrict__ bcat) {
  int n = blockIdx.x * 256 + threadIdx.x;
  int gate = n >> 9, col = n & 511;
  const float* s;
  switch (gate) { case 0: s = b0; break; case 1: s = b1; break;
                  case 2: s = b2; break; default: s = b3; break; }
  bcat[n] = s[col];
}

// ------------------- Phase 1: xw = x @ W + b  (M=32768,K=512,N=2048) -------
// Block = 64x64 output tile, 8 waves, 2 C-tiles (16x16) per wave.
__global__ __launch_bounds__(256) void k_gemm_xw(const bf16* __restrict__ xbf,
                                                 const bf16* __restrict__ Wsw,
                                                 const float* __restrict__ bcat,
                                                 float* __restrict__ xw) {
  int mb = blockIdx.x & 511;            // 512 M blocks
  int nb = blockIdx.x >> 9;             // 32  N blocks
  int wave = threadIdx.x >> 5, lane = threadIdx.x & 31;
  int mt = wave & 3;
  int ntp = (wave >> 2) << 1;           // 0 or 2
  int m0 = mb * 64 + mt * 16;           // global row (b*S+s)
  int nt0 = nb * 4 + ntp;               // global ntile
  const bf16* Ar = xbf + (size_t)m0 * I_;
  const bf16* B0 = Wsw + (size_t)nt0 * (KT * 512);
  const bf16* B1 = B0 + (KT * 512);
  v8f c0 = {0,0,0,0,0,0,0,0}, c1 = {0,0,0,0,0,0,0,0};
  for (int kt = 0; kt < KT; ++kt) {
    v16bf a  = load_a_frag(Ar + kt * 32, I_, lane);
    v16bf b0 = load_b_frag(B0 + kt * 512, lane);
    c0 = wmma_bf16(a, b0, c0);
    v16bf b1 = load_b_frag(B1 + kt * 512, lane);
    c1 = wmma_bf16(a, b1, c1);
  }
  // C layout: elem r -> (m = r + (lane>>4)*8, n = lane&15)
  int cm = m0 + ((lane >> 4) << 3);
  int n0 = nt0 * 16 + (lane & 15);
  float bias0 = bcat[n0], bias1 = bcat[n0 + 16];
  #pragma unroll
  for (int r = 0; r < 8; ++r) {
    int m = cm + r;
    int bb = m >> 9, s = m & 511;        // m = bb*S + s
    float* dst = xw + ((size_t)(s << 6) + bb) * G4;   // xw[s][b][n]
    dst[n0]      = c0[r] + bias0;
    dst[n0 + 16] = c1[r] + bias1;
  }
}

// ------------------- Phase 2: persistent recurrent scan --------------------
// WG wg owns H cols [wg*16, wg*16+16). Gates for those cols = global ntiles
// g*32+wg, g=0..3 (f,i,o,c). 8 waves: wave w -> Mtile w&3, gates {0,1} or {2,3}.
__global__ __launch_bounds__(256) void k_scan(const bf16* __restrict__ Usw,
    const float* __restrict__ xw, float* __restrict__ out,
    bf16* __restrict__ hbf0, bf16* __restrict__ hbf1,
    float* __restrict__ cst, unsigned* __restrict__ bar) {
  __shared__ float gbuf[4][64][17];      // [gate][batch][col], padded
  const int wg = blockIdx.x;
  const int tid = threadIdx.x;
  const int wave = tid >> 5, lane = tid & 31;
  const int mt = wave & 3;
  const int g0 = (wave >> 2) << 1;       // my first gate
  const int g1 = g0 + 1;
  const bf16* Bt0 = Usw + (size_t)(g0 * 32 + wg) * (KT * 512);
  const bf16* Bt1 = Usw + (size_t)(g1 * 32 + wg) * (KT * 512);
  // zero my private cell-state columns (ws is poisoned, not re-poisoned)
  for (int i = tid; i < 1024; i += 256)
    cst[(i >> 4) * H_ + wg * 16 + (i & 15)] = 0.0f;
  const int crow = mt * 16 + ((lane >> 4) << 3);

  for (int s = 0; s < S_; ++s) {
    const bf16* hr = (s & 1) ? hbf1 : hbf0;   // h_t   (read)
    bf16*       hw = (s & 1) ? hbf0 : hbf1;   // h_t+1 (write)
    v8f c0 = {0,0,0,0,0,0,0,0}, c1 = {0,0,0,0,0,0,0,0};
    const bf16* Ar = hr + (size_t)(mt * 16) * H_;
    for (int kt = 0; kt < KT; ++kt) {
      v16bf a  = load_a_frag(Ar + kt * 32, H_, lane);   // h from L2
      v16bf b0 = load_b_frag(Bt0 + kt * 512, lane);     // U from L2
      c0 = wmma_bf16(a, b0, c0);
      v16bf b1 = load_b_frag(Bt1 + kt * 512, lane);
      c1 = wmma_bf16(a, b1, c1);
    }
    #pragma unroll
    for (int r = 0; r < 8; ++r) {
      gbuf[g0][crow + r][lane & 15] = c0[r];
      gbuf[g1][crow + r][lane & 15] = c1[r];
    }
    __syncthreads();
    const float* xws = xw + (size_t)s * (B_ * G4);
    for (int i = tid; i < 1024; i += 256) {
      int m = i >> 4, col = i & 15;
      int hc = wg * 16 + col;
      float gf = gbuf[0][m][col] + xws[m * G4 + 0 * H_ + hc];
      float gi = gbuf[1][m][col] + xws[m * G4 + 1 * H_ + hc];
      float go = gbuf[2][m][col] + xws[m * G4 + 2 * H_ + hc];
      float gc = gbuf[3][m][col] + xws[m * G4 + 3 * H_ + hc];
      float f  = sigmoidf_(gf);
      float ii = sigmoidf_(gi);
      float o  = sigmoidf_(go);
      float ch = tanhf(gc);
      float cn = f * cst[m * H_ + hc] + ii * ch;
      float hn = o * tanhf(cn);
      cst[m * H_ + hc] = cn;
      hw[m * H_ + hc] = (bf16)hn;
      out[((size_t)m * S_ + s) * H_ + hc] = hn;          // result[b][s][h]
      if (s == S_ - 1) {
        out[(size_t)B_ * S_ * H_ + (size_t)m * H_ + hc] = hn;              // h_T
        out[(size_t)B_ * S_ * H_ + (size_t)B_ * H_ + (size_t)m * H_ + hc] = cn; // c_T
      }
    }
    if (s + 1 < S_) {   // pull next step's xw slice toward L2 (64B/thread)
      const float* nf = xw + (size_t)(s + 1) * (B_ * G4) + ((wg << 8) + tid) * 16;
      __builtin_prefetch(nf, 0, 1);
    }
    grid_barrier(bar, (unsigned)(NWG * (s + 1)));
  }
}

// ---------------------------------------------------------------------------
extern "C" void kernel_launch(void* const* d_in, const int* in_sizes, int n_in,
                              void* d_out, int out_size, void* d_ws, size_t ws_size,
                              hipStream_t stream) {
  (void)in_sizes; (void)n_in; (void)out_size; (void)ws_size;
  const float* x   = (const float*)d_in[0];
  const float* Wf  = (const float*)d_in[1];
  const float* Uf  = (const float*)d_in[2];
  const float* bfv = (const float*)d_in[3];
  const float* Wi  = (const float*)d_in[4];
  const float* Ui  = (const float*)d_in[5];
  const float* biv = (const float*)d_in[6];
  const float* Wo  = (const float*)d_in[7];
  const float* Uo  = (const float*)d_in[8];
  const float* bov = (const float*)d_in[9];
  const float* Wc  = (const float*)d_in[10];
  const float* Uc  = (const float*)d_in[11];
  const float* bcv = (const float*)d_in[12];

  char* ws = (char*)d_ws;
  size_t off = 0;
  auto take = [&](size_t bytes) -> char* {
    char* p = ws + off;
    off += (bytes + 255) & ~(size_t)255;
    return p;
  };
  float*    xw   = (float*)take((size_t)S_ * B_ * G4 * 4);   // 256 MB
  bf16*     xbf  = (bf16*) take((size_t)B_ * S_ * I_ * 2);   //  32 MB
  bf16*     Wsw  = (bf16*) take((size_t)I_ * G4 * 2);        //   2 MB
  bf16*     Usw  = (bf16*) take((size_t)H_ * G4 * 2);        //   2 MB
  float*    bcat = (float*)take((size_t)G4 * 4);
  bf16*     h0   = (bf16*) take((size_t)B_ * H_ * 2);
  bf16*     h1   = (bf16*) take((size_t)B_ * H_ * 2);
  float*    cst  = (float*)take((size_t)B_ * H_ * 4);
  unsigned* bar  = (unsigned*)take(256);
  float* out = (float*)d_out;

  k_init<<<dim3(1), dim3(256), 0, stream>>>(h0, bar);
  k_cvt_x<<<dim3((B_ * S_ * I_) / 1024), dim3(256), 0, stream>>>(x, xbf);
  k_cvt_w<<<dim3(1024, 8), dim3(256), 0, stream>>>(Wf, Wi, Wo, Wc,
                                                   Uf, Ui, Uo, Uc, Wsw, Usw);
  k_cvt_b<<<dim3(8), dim3(256), 0, stream>>>(bfv, biv, bov, bcv, bcat);
  k_gemm_xw<<<dim3(512 * 32), dim3(256), 0, stream>>>(xbf, Wsw, bcat, xw);
  k_scan<<<dim3(NWG), dim3(256), 0, stream>>>(Usw, xw, out, h0, h1, cst, bar);
}
